// LTAttention_19782619365623
// MI455X (gfx1250) — compile-verified, matrix operand order checked by reading, not run on hardware
//
#include <hip/hip_runtime.h>
#include <hip/hip_bf16.h>
#include <stdint.h>

typedef __bf16 bf16_t;
typedef __attribute__((ext_vector_type(4)))  __bf16 v4bf;
typedef __attribute__((ext_vector_type(8)))  __bf16 v8bf;
typedef __attribute__((ext_vector_type(16))) __bf16 v16bf;
typedef __attribute__((ext_vector_type(8)))  float  v8f;

#define B_  4
#define S_  1536
#define D_  2048
#define H_  16
#define HD_ 128
#define M_  (B_*S_)

__device__ __forceinline__ bf16_t f2bf(float f) {
  unsigned u = __builtin_bit_cast(unsigned, f);
  unsigned r = (u + 0x7FFFu + ((u >> 16) & 1u)) >> 16;
  unsigned short us = (unsigned short)r;
  return __builtin_bit_cast(bf16_t, us);
}

__device__ __forceinline__ v16bf cat8(v8bf lo, v8bf hi) {
  return __builtin_shufflevector(lo, hi, 0,1,2,3,4,5,6,7,8,9,10,11,12,13,14,15);
}

// CDNA5 async global->LDS copy (16 bytes/lane), tracked on ASYNCcnt.
// LDS operand is the wave-relative LDS byte address = low 32 bits of the
// generic shared pointer.
__device__ __forceinline__ void async_ld_b128(void* lds, const void* gsrc) {
  unsigned l = (unsigned)(unsigned long long)lds;
  unsigned long long g = (unsigned long long)gsrc;
  asm volatile("global_load_async_to_lds_b128 %0, %1, off"
               :: "v"(l), "v"(g) : "memory");
}
__device__ __forceinline__ void wait_asynccnt0() {
  asm volatile("s_wait_asynccnt 0" ::: "memory");
}

// ---------------------------------------------------------------------------
// GEMM: C[M,N] = A[M,K] * W[K,N] + bias[N]   (fp32 in/out, bf16 WMMA compute)
// Block tile 64x128, 8 waves of 32x32, K-step 32.
// ---------------------------------------------------------------------------
__global__ __launch_bounds__(256) void gemm_bf16_wmma(
    const float* __restrict__ A, const float* __restrict__ W,
    const float* __restrict__ bias, float* __restrict__ C,
    int M, int N, int K) {
  __shared__ bf16_t As[64][32];   // M-major A tile (4 KB)
  __shared__ bf16_t Bs[128][32];  // N-major (transposed) W tile (8 KB)

  const int tid  = threadIdx.x;
  const int wave = tid >> 5;
  const int lane = tid & 31;
  const int l15  = lane & 15;
  const int h4   = lane >> 4;
  const int mwave = (wave >> 2) * 32;  // 0 / 32
  const int nwave = (wave & 3) * 32;   // 0,32,64,96
  const int mbase = blockIdx.x * 64;
  const int nbase = blockIdx.y * 128;

  v8f acc[2][2] = {};

  const int am = tid >> 3, akq = (tid & 7) * 4;        // A staging coords
  const int bkk = tid >> 5, bnq = (tid & 31) * 4;      // B staging coords

  for (int k0 = 0; k0 < K; k0 += 32) {
    __syncthreads();
    // ---- stage A tile (fp32 -> bf16) ----
    #pragma unroll
    for (int rep = 0; rep < 2; ++rep) {
      const float4 f = *(const float4*)(A + (size_t)(mbase + am + rep * 32) * K + k0 + akq);
      v4bf p; p[0] = f2bf(f.x); p[1] = f2bf(f.y); p[2] = f2bf(f.z); p[3] = f2bf(f.w);
      *(v4bf*)&As[am + rep * 32][akq] = p;
    }
    // ---- stage B tile transposed (fp32 -> bf16) ----
    #pragma unroll
    for (int rep = 0; rep < 4; ++rep) {
      const int kkr = bkk + rep * 8;
      const float4 f = *(const float4*)(W + (size_t)(k0 + kkr) * N + nbase + bnq);
      Bs[bnq + 0][kkr] = f2bf(f.x);
      Bs[bnq + 1][kkr] = f2bf(f.y);
      Bs[bnq + 2][kkr] = f2bf(f.z);
      Bs[bnq + 3][kkr] = f2bf(f.w);
    }
    __syncthreads();

    v16bf af[2], bf[2];
    #pragma unroll
    for (int mt = 0; mt < 2; ++mt) {
      const bf16_t* p = &As[mwave + mt * 16 + l15][h4 * 8];
      af[mt] = cat8(*(const v8bf*)p, *(const v8bf*)(p + 16));
    }
    #pragma unroll
    for (int nt = 0; nt < 2; ++nt) {
      const bf16_t* p = &Bs[nwave + nt * 16 + l15][h4 * 8];
      bf[nt] = cat8(*(const v8bf*)p, *(const v8bf*)(p + 16));
    }
    #pragma unroll
    for (int mt = 0; mt < 2; ++mt)
      #pragma unroll
      for (int nt = 0; nt < 2; ++nt)
        acc[mt][nt] = __builtin_amdgcn_wmma_f32_16x16x32_bf16(
            false, af[mt], false, bf[nt], (short)0, acc[mt][nt], false, false);
  }

  #pragma unroll
  for (int mt = 0; mt < 2; ++mt)
    #pragma unroll
    for (int nt = 0; nt < 2; ++nt) {
      const int col = nbase + nwave + nt * 16 + l15;
      const float bv = bias[col];
      #pragma unroll
      for (int r = 0; r < 8; ++r) {
        const int row = mbase + mwave + mt * 16 + r + 8 * h4;
        C[(size_t)row * N + col] = acc[mt][nt][r] + bv;
      }
    }
}

// ---------------------------------------------------------------------------
// RMSNorm over D for q,k; pack q,k -> bf16 [B,H,S,hd]; v -> bf16 [B,H,hd,S]
// ---------------------------------------------------------------------------
__global__ __launch_bounds__(256) void rmsnorm_pack_kernel(
    const float* __restrict__ qkv, const float* __restrict__ qw,
    const float* __restrict__ kw, bf16_t* __restrict__ qb,
    bf16_t* __restrict__ kb, bf16_t* __restrict__ vtb) {
  const int row = blockIdx.x;          // b*S + s
  const int b = row / S_, s = row - b * S_;
  const int tid = threadIdx.x;
  const float* qr = qkv + (size_t)row * (3 * D_);
  const float* kr = qr + D_;
  const float* vr = qr + 2 * D_;

  float qv[8], kv[8], vv[8];
  float sq = 0.f, sk = 0.f;
  #pragma unroll
  for (int i = 0; i < 8; ++i) {
    const int d = tid + i * 256;
    qv[i] = qr[d]; kv[i] = kr[d]; vv[i] = vr[d];
    sq += qv[i] * qv[i];
    sk += kv[i] * kv[i];
  }
  __shared__ float r1[256], r2[256];
  r1[tid] = sq; r2[tid] = sk;
  __syncthreads();
  for (int st = 128; st > 0; st >>= 1) {
    if (tid < st) { r1[tid] += r1[tid + st]; r2[tid] += r2[tid + st]; }
    __syncthreads();
  }
  const float rq = rsqrtf(r1[0] * (1.0f / D_) + 1e-6f);
  const float rk = rsqrtf(r2[0] * (1.0f / D_) + 1e-6f);

  #pragma unroll
  for (int i = 0; i < 8; ++i) {
    const int d = tid + i * 256;
    const int hh = d >> 7, c = d & 127;
    const size_t qo = ((size_t)(b * H_ + hh) * S_ + s) * HD_ + c;
    qb[qo] = f2bf(qv[i] * rq * qw[d]);
    kb[qo] = f2bf(kv[i] * rk * kw[d]);
    vtb[((size_t)(b * H_ + hh) * HD_ + c) * S_ + s] = f2bf(vv[i]);
  }
}

// ---------------------------------------------------------------------------
// Causal flash attention. Block = 4 waves = 64 q rows of one (b,h).
// Wave owns 16 q rows; K/V staged in 32-key LDS tiles via ASYNC global->LDS;
// online softmax; PV via WMMA with alpha rescale.
// ---------------------------------------------------------------------------
__global__ __launch_bounds__(128) void attn_kernel(
    const bf16_t* __restrict__ qb, const bf16_t* __restrict__ kb,
    const bf16_t* __restrict__ vtb, float* __restrict__ attn) {
  __shared__ bf16_t Kt[32][128];      // [key][hd]  8 KB
  __shared__ bf16_t Vt[128][32];      // [hd][key]  8 KB (pre-transposed V)
  __shared__ bf16_t Pw[4][16][32];    // per-wave P scratch 4 KB

  const int nq    = S_ / 64;
  const int bhead = blockIdx.x / nq;  // b*H + h
  const int qt    = blockIdx.x % nq;
  const int qbase = qt * 64;
  const int tid = threadIdx.x, wave = tid >> 5, lane = tid & 31;
  const int l15 = lane & 15, h4 = lane >> 4;
  const int qrow0 = qbase + wave * 16;
  const int qmax  = qrow0 + 15;
  const float scale = 0.08838834764831845f;  // 1/sqrt(128)

  const bf16_t* Qbase = qb  + (size_t)bhead * S_ * HD_;
  const bf16_t* Kbase = kb  + (size_t)bhead * S_ * HD_;
  const bf16_t* Vbase = vtb + (size_t)bhead * HD_ * S_;

  // resident Q fragments (16 x 128 as 4 chunks of K=32)
  v16bf qf[4];
  {
    const int qr = qrow0 + l15;
    #pragma unroll
    for (int c = 0; c < 4; ++c) {
      const bf16_t* p = Qbase + (size_t)qr * HD_ + c * 32 + h4 * 8;
      qf[c] = cat8(*(const v8bf*)p, *(const v8bf*)(p + 16));
    }
  }

  v8f o[8] = {};
  float mrow[8], lrow[8];
  #pragma unroll
  for (int r = 0; r < 8; ++r) { mrow[r] = -1e30f; lrow[r] = 0.f; }

  // per-thread staging coordinates (512 x 16B chunks per tile, 128 threads)
  const int kc_row[4] = { (tid + 0)   >> 4, (tid + 128) >> 4,
                          (tid + 256) >> 4, (tid + 384) >> 4 };
  const int vc_row[4] = { (tid + 0)   >> 2, (tid + 128) >> 2,
                          (tid + 256) >> 2, (tid + 384) >> 2 };

  const int nkb = (qbase + 64) >> 5;
  for (int kbi = 0; kbi < nkb; ++kbi) {
    const int kstart = kbi * 32;
    __syncthreads();
    // ---- async stage K tile [32][128] : 4 x b128 per thread ----
    #pragma unroll
    for (int i = 0; i < 4; ++i) {
      const int c = tid + i * 128;
      const int krow = c >> 4, off = (c & 15) * 8;
      async_ld_b128(&Kt[krow][off],
                    Kbase + (size_t)(kstart + krow) * HD_ + off);
    }
    // ---- async stage V^T tile [128][32] : 4 x b128 per thread ----
    #pragma unroll
    for (int i = 0; i < 4; ++i) {
      const int c = tid + i * 128;
      const int vrow = c >> 2, off = (c & 3) * 8;
      async_ld_b128(&Vt[vrow][off],
                    Vbase + (size_t)vrow * S_ + kstart + off);
    }
    // prefetch next tiles into cache while async copies drain
    if (kbi + 1 < nkb) {
      __builtin_prefetch(Kbase + (size_t)(kstart + 32 + kc_row[0]) * HD_, 0, 1);
      __builtin_prefetch(Vbase + (size_t)vc_row[0] * S_ + kstart + 32, 0, 1);
    }
    wait_asynccnt0();
    __syncthreads();

    if (kstart <= qmax) {                  // wave-uniform predicate
      // scores: two 16x16 tiles over 32 keys
      v8f sc[2] = {};
      #pragma unroll
      for (int t = 0; t < 2; ++t)
        #pragma unroll
        for (int c = 0; c < 4; ++c) {
          const bf16_t* p = &Kt[t * 16 + l15][c * 32 + h4 * 8];
          v16bf kf = cat8(*(const v8bf*)p, *(const v8bf*)(p + 16));
          sc[t] = __builtin_amdgcn_wmma_f32_16x16x32_bf16(
              false, qf[c], false, kf, (short)0, sc[t], false, false);
        }

      // causal mask + online softmax (row stats per accum index r)
      float pn0[8], pn1[8], alpha[8];
      #pragma unroll
      for (int r = 0; r < 8; ++r) {
        const int qi = qrow0 + r + 8 * h4;
        float v0 = sc[0][r] * scale;
        float v1 = sc[1][r] * scale;
        if (kstart + l15 > qi)      v0 = -1e30f;
        if (kstart + 16 + l15 > qi) v1 = -1e30f;
        float mx = fmaxf(v0, v1);
        #pragma unroll
        for (int msk = 1; msk < 16; msk <<= 1)
          mx = fmaxf(mx, __shfl_xor(mx, msk, 32));
        const float mnew = fmaxf(mrow[r], mx);
        const float a  = __expf(mrow[r] - mnew);
        const float p0 = __expf(v0 - mnew);
        const float p1 = __expf(v1 - mnew);
        float srow = p0 + p1;
        #pragma unroll
        for (int msk = 1; msk < 16; msk <<= 1)
          srow += __shfl_xor(srow, msk, 32);
        lrow[r] = lrow[r] * a + srow;
        mrow[r] = mnew;
        alpha[r] = a;
        pn0[r] = p0; pn1[r] = p1;
      }

      // reshape P into an A-fragment via per-wave LDS
      #pragma unroll
      for (int r = 0; r < 8; ++r) {
        Pw[wave][r + 8 * h4][l15]      = f2bf(pn0[r]);
        Pw[wave][r + 8 * h4][16 + l15] = f2bf(pn1[r]);
      }
      v16bf pf;
      {
        const bf16_t* p = &Pw[wave][l15][h4 * 8];
        pf = cat8(*(const v8bf*)p, *(const v8bf*)(p + 16));
      }

      // O(16x128) += P(16x32) * V(32x128), with alpha rescale
      #pragma unroll
      for (int ct = 0; ct < 8; ++ct) {
        const bf16_t* pv = &Vt[ct * 16 + l15][h4 * 8];
        v16bf vf = cat8(*(const v8bf*)pv, *(const v8bf*)(pv + 16));
        #pragma unroll
        for (int r = 0; r < 8; ++r) o[ct][r] *= alpha[r];
        o[ct] = __builtin_amdgcn_wmma_f32_16x16x32_bf16(
            false, pf, false, vf, (short)0, o[ct], false, false);
      }
    }
  }

  // normalize and write [B,S,D] fp32
  float rl[8];
  #pragma unroll
  for (int r = 0; r < 8; ++r) rl[r] = 1.0f / lrow[r];
  const int b = bhead >> 4, h = bhead & 15;
  #pragma unroll
  for (int ct = 0; ct < 8; ++ct) {
    const int col = h * HD_ + ct * 16 + l15;
    #pragma unroll
    for (int r = 0; r < 8; ++r) {
      const int row = b * S_ + qrow0 + r + 8 * h4;
      attn[(size_t)row * D_ + col] = o[ct][r] * rl[r];
    }
  }
}

// ---------------------------------------------------------------------------
extern "C" void kernel_launch(void* const* d_in, const int* in_sizes, int n_in,
                              void* d_out, int out_size, void* d_ws, size_t ws_size,
                              hipStream_t stream) {
  const float* x      = (const float*)d_in[0];
  const float* w_qkv  = (const float*)d_in[1];
  const float* b_qkv  = (const float*)d_in[2];
  const float* q_ln_w = (const float*)d_in[3];
  const float* k_ln_w = (const float*)d_in[4];
  const float* w_out  = (const float*)d_in[5];
  const float* b_out  = (const float*)d_in[6];
  float* out = (float*)d_out;

  char* w = (char*)d_ws;
  size_t off = 0;
  float*  qkv  = (float*)(w + off); off += (size_t)M_ * (3 * D_) * sizeof(float);
  bf16_t* qb   = (bf16_t*)(w + off); off += (size_t)M_ * D_ * sizeof(bf16_t);
  bf16_t* kb   = (bf16_t*)(w + off); off += (size_t)M_ * D_ * sizeof(bf16_t);
  bf16_t* vtb  = (bf16_t*)(w + off); off += (size_t)M_ * D_ * sizeof(bf16_t);
  float*  attn = (float*)(w + off); off += (size_t)M_ * D_ * sizeof(float);
  (void)ws_size; (void)in_sizes; (void)n_in; (void)out_size;

  // 1) QKV projection
  gemm_bf16_wmma<<<dim3(M_ / 64, (3 * D_) / 128), dim3(256), 0, stream>>>(
      x, w_qkv, b_qkv, qkv, M_, 3 * D_, D_);
  // 2) QK-RMSNorm + bf16 packing (V transposed per head)
  rmsnorm_pack_kernel<<<dim3(M_), dim3(256), 0, stream>>>(
      qkv, q_ln_w, k_ln_w, qb, kb, vtb);
  // 3) causal flash attention
  attn_kernel<<<dim3(B_ * H_ * (S_ / 64)), dim3(128), 0, stream>>>(
      qb, kb, vtb, attn);
  // 4) output projection
  gemm_bf16_wmma<<<dim3(M_ / 64, D_ / 128), dim3(256), 0, stream>>>(
      attn, w_out, b_out, out, M_, D_, D_);
}
